// BitNetMLP_19421842112750
// MI455X (gfx1250) — matrix operand order repaired and checked by reading, not compiled, not run on hardware
//
#include <hip/hip_runtime.h>
#include <hip/hip_bf16.h>
#include <stdint.h>

// ---------------------------------------------------------------------------
// BitNet MLP for MI455X (gfx1250): bf16 WMMA GEMMs with in-register ternary
// weight decode, async global->LDS A-tile staging (ASYNCcnt), single barrier
// per K-step with double-buffered LDS, fused ReLU^2 gating + RMSNorm.
// ---------------------------------------------------------------------------

typedef __attribute__((ext_vector_type(16))) __bf16 v16bf;
typedef __attribute__((ext_vector_type(8)))  float  v8f;

#define H_DIM 2560
#define I_DIM 6912
#define M_TOK 8192
#define RMS_EPS 1e-6f

// Issue two async 16B global->LDS copies (32 bytes per thread).
// INST_OFFSET is added to BOTH the LDS and global address (ISA 08 §4.4).
#define ASYNC_A2(ldsaddr, gptr)                                        \
  asm volatile("global_load_async_to_lds_b128 %0, %1, off\n\t"         \
               "global_load_async_to_lds_b128 %0, %1, off offset:16"   \
               :: "v"(ldsaddr), "v"(gptr) : "memory")

#define WAIT_ASYNC0() asm volatile("s_wait_asynccnt 0x0" ::: "memory")

// low 32 bits of a generic LDS pointer == LDS byte offset (ISA §10.2)
static __device__ __forceinline__ unsigned lds_off(const void* p) {
  return (unsigned)(uintptr_t)p;
}

// float -> bf16 (round to nearest even)
static __device__ __forceinline__ unsigned short f2bf(float f) {
  unsigned u = __builtin_bit_cast(unsigned, f);
  u += 0x7FFFu + ((u >> 16) & 1u);
  return (unsigned short)(u >> 16);
}

struct U128x2 { uint4 lo, hi; };

static __device__ __forceinline__ v16bf load_frag(const unsigned short* p0,
                                                  const unsigned short* p1) {
  U128x2 t;
  t.lo = *(const uint4*)p0;
  t.hi = *(const uint4*)p1;
  return __builtin_bit_cast(v16bf, t);
}

static __device__ __forceinline__ v8f wmma_bf16(v16bf a, v16bf b, v8f c) {
  return __builtin_amdgcn_wmma_f32_16x16x32_bf16(
      /*neg_a=*/false, a, /*neg_b=*/false, b,
      /*c_mod=*/(short)0, c, /*reuse_a=*/false, /*reuse_b=*/false);
}

// Decode 8 packed bytes (one fixed 2-bit field each) -> 8 bf16 weights
// {-1,0,1,2} packed as uint4 (little-endian ushort order).
static __device__ __forceinline__ uint4 decode8(uint2 raw, int shift) {
  const unsigned long long LUT = 0x40003F800000BF80ull; // [2,1,0,-1] bf16 bits
  unsigned b[8];
#pragma unroll
  for (int i = 0; i < 4; ++i) b[i]     = (raw.x >> (8 * i)) & 0xFFu;
#pragma unroll
  for (int i = 0; i < 4; ++i) b[4 + i] = (raw.y >> (8 * i)) & 0xFFu;
  unsigned w[8];
#pragma unroll
  for (int i = 0; i < 8; ++i) {
    unsigned q = (b[i] >> shift) & 3u;
    w[i] = (unsigned)((LUT >> (q * 16)) & 0xFFFFull);
  }
  uint4 out;
  out.x = w[0] | (w[1] << 16);
  out.y = w[2] | (w[3] << 16);
  out.z = w[4] | (w[5] << 16);
  out.w = w[6] | (w[7] << 16);
  return out;
}

// ---------------------------------------------------------------------------
// Kernel 1: x fp32 -> bf16
// ---------------------------------------------------------------------------
__global__ __launch_bounds__(256)
void bitnet_cvt_bf16(const float* __restrict__ x,
                     unsigned short* __restrict__ xbf) {
  int i = (blockIdx.x * 256 + threadIdx.x) * 8;
  float4 f0 = ((const float4*)(x + i))[0];
  float4 f1 = ((const float4*)(x + i))[1];
  uint4 o;
  o.x = (unsigned)f2bf(f0.x) | ((unsigned)f2bf(f0.y) << 16);
  o.y = (unsigned)f2bf(f0.z) | ((unsigned)f2bf(f0.w) << 16);
  o.z = (unsigned)f2bf(f1.x) | ((unsigned)f2bf(f1.y) << 16);
  o.w = (unsigned)f2bf(f1.z) | ((unsigned)f2bf(f1.w) << 16);
  *(uint4*)(xbf + i) = o;
}

// ---------------------------------------------------------------------------
// Kernel 2: fused gate+up GEMM + ReLU^2 gating + RMS statistic.
// WG = 256 threads = 8 waves (4 along M, 2 along N). Tile 128(M) x 64(N),
// K step 32, double-buffered LDS, async A staging, one barrier per step.
// ---------------------------------------------------------------------------
__global__ __launch_bounds__(256)
void bitnet_gateup(const unsigned short* __restrict__ xbf,
                   const unsigned char* __restrict__ gpack,
                   const unsigned char* __restrict__ upack,
                   const float* __restrict__ gscale_p,
                   const float* __restrict__ uscale_p,
                   const float* __restrict__ norm_w,
                   unsigned short* __restrict__ hidden,
                   float* __restrict__ sumsq) {
  __shared__ unsigned short Alds[2][128 * 40];
  __shared__ unsigned short Bg[2][64 * 40];
  __shared__ unsigned short Bu[2][64 * 40];

  const int tid  = threadIdx.x;
  const int lane = tid & 31;
  const int wid  = tid >> 5;
  const int l16  = lane & 15;
  const int half = lane >> 4;
  const int wm   = wid & 3;
  const int wn   = wid >> 2;

  const int m0 = blockIdx.y * 128;
  const int n0 = blockIdx.x * 64;

  v8f zero = {};
  v8f accg[2][2], accu[2][2];
#pragma unroll
  for (int i = 0; i < 2; ++i)
#pragma unroll
    for (int j = 0; j < 2; ++j) { accg[i][j] = zero; accu[i][j] = zero; }

  // cooperative-load indices
  const int lrow = tid >> 1;       // 0..127 A row
  const int lseg = tid & 1;        // 0..1   A 16-element segment
  const int bcol = tid & 63;       // 0..63  B column
  const int bq   = tid >> 6;       // 0..3   B 8-byte chunk
  const int KB   = H_DIM / 4;
  const int NT   = H_DIM / 32;

  const unsigned short* aglob =
      xbf + (size_t)(m0 + lrow) * H_DIM + lseg * 16;
  const unsigned la[2] = {
      lds_off(&Alds[0][lrow * 40 + lseg * 16]),
      lds_off(&Alds[1][lrow * 40 + lseg * 16])};
  const unsigned char* gsrc = gpack + (size_t)(n0 + bcol) * KB + bq * 8;
  const unsigned char* usrc = upack + (size_t)(n0 + bcol) * KB + bq * 8;

  // prologue: tile 0 in flight
  ASYNC_A2(la[0], aglob);
  uint2 graw = *(const uint2*)gsrc;
  uint2 uraw = *(const uint2*)usrc;

  for (int it = 0; it < NT; ++it) {
    const int buf = it & 1;

    // decode current B tiles into LDS
    const int shift = 6 - 2 * (it & 3);
    *(uint4*)&Bg[buf][bcol * 40 + bq * 8] = decode8(graw, shift);
    *(uint4*)&Bu[buf][bcol * 40 + bq * 8] = decode8(uraw, shift);

    // register-prefetch next packed B bytes
    if (it + 1 < NT) {
      const size_t nb = (size_t)((it + 1) >> 2) * 32;
      graw = *(const uint2*)(gsrc + nb);
      uraw = *(const uint2*)(usrc + nb);
    }

    WAIT_ASYNC0();        // A(it) landed in Alds[buf]
    __syncthreads();      // (compiler adds s_wait_dscnt 0 before signal)

    // issue A(it+1) AFTER the barrier: no wave can still be reading buf^1
    if (it + 1 < NT) ASYNC_A2(la[buf ^ 1], aglob + (it + 1) * 32);

    const unsigned short* Ab  = Alds[buf];
    const unsigned short* Bgb = Bg[buf];
    const unsigned short* Bub = Bu[buf];

    v16bf afrag[2], bgf[2], buf_[2];
#pragma unroll
    for (int sm = 0; sm < 2; ++sm) {
      const unsigned short* p = &Ab[(wm * 32 + sm * 16 + l16) * 40 + half * 8];
      afrag[sm] = load_frag(p, p + 16);
    }
#pragma unroll
    for (int sn = 0; sn < 2; ++sn) {
      const unsigned short* pg = &Bgb[(wn * 32 + sn * 16 + l16) * 40 + half * 8];
      const unsigned short* pu = &Bub[(wn * 32 + sn * 16 + l16) * 40 + half * 8];
      bgf[sn]  = load_frag(pg, pg + 16);
      buf_[sn] = load_frag(pu, pu + 16);
    }
#pragma unroll
    for (int sm = 0; sm < 2; ++sm)
#pragma unroll
      for (int sn = 0; sn < 2; ++sn) {
        accg[sm][sn] = wmma_bf16(afrag[sm], bgf[sn],  accg[sm][sn]);
        accu[sm][sn] = wmma_bf16(afrag[sm], buf_[sn], accu[sm][sn]);
      }
  }

  // --- epilogue: gating, norm_w fold, bf16 store, RMS sum-of-squares ---
  const float gs = *gscale_p;
  const float us = *uscale_p;

#pragma unroll
  for (int sm = 0; sm < 2; ++sm) {
    float ss[8];
#pragma unroll
    for (int v = 0; v < 8; ++v) ss[v] = 0.0f;

#pragma unroll
    for (int sn = 0; sn < 2; ++sn) {
      const int C = n0 + wn * 32 + sn * 16 + l16;
      const float nw = norm_w[C];
#pragma unroll
      for (int v = 0; v < 8; ++v) {
        float g = accg[sm][sn][v] * gs;
        float u = accu[sm][sn][v] * us;
        float r = fmaxf(g, 0.0f);
        float h = r * r * u;            // relu(gate)^2 * up
        ss[v] += h * h;
        const int R = m0 + wm * 32 + sm * 16 + half * 8 + v;
        hidden[(size_t)R * I_DIM + C] = f2bf(h * nw);
      }
    }
#pragma unroll
    for (int v = 0; v < 8; ++v) {
      float t = ss[v];
      t += __shfl_xor(t, 1, 32);
      t += __shfl_xor(t, 2, 32);
      t += __shfl_xor(t, 4, 32);
      t += __shfl_xor(t, 8, 32);
      if (l16 == 0) {
        const int R = m0 + wm * 32 + sm * 16 + half * 8 + v;
        atomicAdd(&sumsq[R], t);
      }
    }
  }
}

// ---------------------------------------------------------------------------
// Kernel 3: per-row rsqrt factor
// ---------------------------------------------------------------------------
__global__ __launch_bounds__(256)
void bitnet_rms(const float* __restrict__ sumsq, float* __restrict__ srow) {
  int r = blockIdx.x * 256 + threadIdx.x;
  srow[r] = rsqrtf(sumsq[r] * (1.0f / (float)I_DIM) + RMS_EPS);
}

// ---------------------------------------------------------------------------
// Kernel 4: down GEMM (M x H, K = I) with per-row RMS scale.
// ---------------------------------------------------------------------------
__global__ __launch_bounds__(256)
void bitnet_down(const unsigned short* __restrict__ hidden,
                 const unsigned char* __restrict__ dpack,
                 const float* __restrict__ dscale_p,
                 const float* __restrict__ srow,
                 float* __restrict__ out) {
  __shared__ unsigned short Alds[2][128 * 40];
  __shared__ unsigned short Bd[2][64 * 40];

  const int tid  = threadIdx.x;
  const int lane = tid & 31;
  const int wid  = tid >> 5;
  const int l16  = lane & 15;
  const int half = lane >> 4;
  const int wm   = wid & 3;
  const int wn   = wid >> 2;

  const int m0 = blockIdx.y * 128;
  const int n0 = blockIdx.x * 64;

  v8f zero = {};
  v8f acc[2][2];
#pragma unroll
  for (int i = 0; i < 2; ++i)
#pragma unroll
    for (int j = 0; j < 2; ++j) acc[i][j] = zero;

  const int lrow = tid >> 1;
  const int lseg = tid & 1;
  const int bcol = tid & 63;
  const int bq   = tid >> 6;
  const int KB   = I_DIM / 4;
  const int NT   = I_DIM / 32;

  const unsigned short* aglob =
      hidden + (size_t)(m0 + lrow) * I_DIM + lseg * 16;
  const unsigned la[2] = {
      lds_off(&Alds[0][lrow * 40 + lseg * 16]),
      lds_off(&Alds[1][lrow * 40 + lseg * 16])};
  const unsigned char* dsrc = dpack + (size_t)(n0 + bcol) * KB + bq * 8;

  ASYNC_A2(la[0], aglob);
  uint2 draw = *(const uint2*)dsrc;

  for (int it = 0; it < NT; ++it) {
    const int buf = it & 1;

    const int shift = 6 - 2 * (it & 3);
    *(uint4*)&Bd[buf][bcol * 40 + bq * 8] = decode8(draw, shift);

    if (it + 1 < NT) {
      const size_t nb = (size_t)((it + 1) >> 2) * 32;
      draw = *(const uint2*)(dsrc + nb);
    }

    WAIT_ASYNC0();
    __syncthreads();

    if (it + 1 < NT) ASYNC_A2(la[buf ^ 1], aglob + (it + 1) * 32);

    const unsigned short* Ab  = Alds[buf];
    const unsigned short* Bdb = Bd[buf];

    v16bf afrag[2], bdf[2];
#pragma unroll
    for (int sm = 0; sm < 2; ++sm) {
      const unsigned short* p = &Ab[(wm * 32 + sm * 16 + l16) * 40 + half * 8];
      afrag[sm] = load_frag(p, p + 16);
    }
#pragma unroll
    for (int sn = 0; sn < 2; ++sn) {
      const unsigned short* pb = &Bdb[(wn * 32 + sn * 16 + l16) * 40 + half * 8];
      bdf[sn] = load_frag(pb, pb + 16);
    }
#pragma unroll
    for (int sm = 0; sm < 2; ++sm)
#pragma unroll
      for (int sn = 0; sn < 2; ++sn)
        acc[sm][sn] = wmma_bf16(afrag[sm], bdf[sn], acc[sm][sn]);
  }

  const float dsc = *dscale_p;
#pragma unroll
  for (int sm = 0; sm < 2; ++sm)
#pragma unroll
    for (int sn = 0; sn < 2; ++sn) {
      const int C = n0 + wn * 32 + sn * 16 + l16;
#pragma unroll
      for (int v = 0; v < 8; ++v) {
        const int R = m0 + wm * 32 + sm * 16 + half * 8 + v;
        out[(size_t)R * H_DIM + C] = acc[sm][sn][v] * dsc * srow[R];
      }
    }
}

// ---------------------------------------------------------------------------
// Host-side launcher
// ---------------------------------------------------------------------------
extern "C" void kernel_launch(void* const* d_in, const int* in_sizes, int n_in,
                              void* d_out, int out_size, void* d_ws, size_t ws_size,
                              hipStream_t stream) {
  const float*         x      = (const float*)d_in[0];
  const unsigned char* gpack  = (const unsigned char*)d_in[1];
  const float*         gscale = (const float*)d_in[2];
  const unsigned char* upack  = (const unsigned char*)d_in[3];
  const float*         uscale = (const float*)d_in[4];
  const unsigned char* dpack  = (const unsigned char*)d_in[5];
  const float*         dscale = (const float*)d_in[6];
  const float*         norm_w = (const float*)d_in[7];
  float*               out    = (float*)d_out;

  char* ws = (char*)d_ws;
  const size_t xbf_bytes    = (size_t)M_TOK * H_DIM * 2;  // 41,943,040
  const size_t hidden_bytes = (size_t)M_TOK * I_DIM * 2;  // 113,246,208
  unsigned short* xbf    = (unsigned short*)ws;
  unsigned short* hidden = (unsigned short*)(ws + xbf_bytes);
  float* sumsq = (float*)(ws + xbf_bytes + hidden_bytes);
  float* srow  = (float*)(ws + xbf_bytes + hidden_bytes + (size_t)M_TOK * 4);

  const int nx = M_TOK * H_DIM;
  bitnet_cvt_bf16<<<nx / (256 * 8), 256, 0, stream>>>(x, xbf);

  hipMemsetAsync(sumsq, 0, (size_t)M_TOK * sizeof(float), stream);

  bitnet_gateup<<<dim3(I_DIM / 64, M_TOK / 128), 256, 0, stream>>>(
      xbf, gpack, upack, gscale, uscale, norm_w, hidden, sumsq);

  bitnet_rms<<<M_TOK / 256, 256, 0, stream>>>(sumsq, srow);

  bitnet_down<<<dim3(H_DIM / 64, M_TOK / 128), 256, 0, stream>>>(
      hidden, dpack, dscale, srow, out);
}
